// CrossViewSpatialFusion_64836826300947
// MI455X (gfx1250) — matrix operand order repaired and checked by reading, not compiled, not run on hardware
//
#include <hip/hip_runtime.h>
#include <hip/hip_bf16.h>

typedef __attribute__((ext_vector_type(16))) _Float16 v16h;
typedef __attribute__((ext_vector_type(8)))  _Float16 v8h;
typedef __attribute__((ext_vector_type(8)))  float    v8f;
typedef _Float16 h16;

#define HW    2304      // 48*48
#define W48   48
#define CIN   64
#define HPA   192       // hidden per rate; also 3*HIDP
#define KPA   576       // 9 taps * 64 ic (tap-major)
#define NBN   12        // B*N
#define KVHW  256       // pooled tokens
#define GT    144       // 2304/16 position tiles

// ---------------------------------------------------------------------------
// WMMA helper: D = A(16x32 f16) * B(32x16 f16) + C(16x16 f32)
// ---------------------------------------------------------------------------
__device__ __forceinline__ v8f wmma16(v16h a, v16h b, v8f c) {
  return __builtin_amdgcn_wmma_f32_16x16x32_f16(false, a, false, b, (short)0, c,
                                                false, false);
}

__device__ __forceinline__ v16h cat8(v8h lo, v8h hi) {
  return __builtin_shufflevector(lo, hi, 0, 1, 2, 3, 4, 5, 6, 7,
                                 8, 9, 10, 11, 12, 13, 14, 15);
}

// A fragment, row-major source A[m][k] (lda mult of 8). Lane g*16+m' holds
// K = 16*g + i  -> two 16B loads.
__device__ __forceinline__ v16h loadA_rm(const h16* __restrict__ A, int lda,
                                         int m0, int k0, int lane) {
  int m = m0 + (lane & 15);
  int g = lane >> 4;
  const v8h* p = (const v8h*)(A + (size_t)m * lda + k0 + 16 * g);
  return cat8(p[0], p[1]);
}

// B fragment with contiguous K: B[k][n] = S[n*ld + k]. Lane kg*16+n holds
// K = 16*kg + i -> two 16B loads.
__device__ __forceinline__ v16h loadB_ctg(const h16* __restrict__ S, int ld,
                                          int k0, int n0, int lane) {
  int n = n0 + (lane & 15);
  int g = lane >> 4;
  const v8h* p = (const v8h*)(S + (size_t)n * ld + k0 + 16 * g);
  return cat8(p[0], p[1]);
}

// im2col B fragment, pos-major input [pos][ldc], tap-major K (k = tap*64+ic):
// a 16-run stays within one tap -> one bounds check + two 16B loads.
__device__ __forceinline__ v16h loadB_im2col(const h16* __restrict__ xc, int ldc,
                                             int k0, int pos0, int rate, int lane) {
  int n = pos0 + (lane & 15);
  int g = lane >> 4;
  int y = n / W48, x = n % W48;
  int k   = k0 + 16 * g;
  int tap = k >> 6;          // 0..8
  int ic0 = k & 63;
  int sy = y + rate * (tap / 3 - 1);
  int sx = x + rate * (tap % 3 - 1);
  v8h lo = {}, hi = {};
  if (sy >= 0 && sy < W48 && sx >= 0 && sx < W48) {
    const v8h* p = (const v8h*)(xc + ((size_t)sy * W48 + sx) * ldc + ic0);
    lo = p[0];
    hi = p[1];
  }
  return cat8(lo, hi);
}

// Cooperative stage of a [rows x 32] f16 weight tile into LDS (dst stride 32).
// src row stride = src_ld halves; 16B chunks, 4 per row.
__device__ __forceinline__ void stage_wtile(const h16* __restrict__ src, int src_ld,
                                            int rows, h16* dst, int tid) {
  int chunks = rows * 4;
  for (int c = tid; c < chunks; c += 128) {
    int row = c >> 2;
    int ko  = (c & 3) * 8;
    *(v8h*)(dst + row * 32 + ko) = *(const v8h*)(src + (size_t)row * src_ld + ko);
  }
}

// ---------------------------------------------------------------------------
// staging kernels
// ---------------------------------------------------------------------------
__global__ void k_cvt_f16(const float* __restrict__ src, h16* __restrict__ dst, int n) {
  int i = blockIdx.x * blockDim.x + threadIdx.x;
  if (i < n) dst[i] = (h16)src[i];
}

// x [bn][64][2304] f32 -> xh [bn][2304][64] f16 (pos-major)
__global__ void k_transpose_x(const float* __restrict__ x, h16* __restrict__ xh) {
  int i = blockIdx.x * blockDim.x + threadIdx.x;
  if (i >= NBN * HW * CIN) return;
  int ch  = i & 63;
  int pos = (i >> 6) % HW;
  int bn  = i / (HW * CIN);
  xh[i] = (h16)x[((size_t)bn * CIN + ch) * HW + pos];
}

// repack 3x3 conv weights: src f32 [oc][ic*9+tap] -> dst f16 [oc][tap*64+ic]
__global__ void k_pack_w(const float* __restrict__ src, h16* __restrict__ dst, int noc) {
  int i = blockIdx.x * blockDim.x + threadIdx.x;
  if (i >= noc * KPA) return;
  int k  = i % KPA;
  int oc = i / KPA;
  int tap = k >> 6;
  int ic  = k & 63;
  dst[i] = (h16)src[(size_t)oc * KPA + ic * 9 + tap];
}

// ---------------------------------------------------------------------------
// K1: three dilated 3x3 convs 64->192 (rates 1..3), +bias, no relu.
// Block of 4 waves shares (bn, rate) and the LDS-staged weight k-tile.
// ---------------------------------------------------------------------------
__global__ void __launch_bounds__(128)
k_dilconv(const h16* __restrict__ xh, const h16* __restrict__ wpa,
          const float* __restrict__ b1, const float* __restrict__ b2,
          const float* __restrict__ b3, h16* __restrict__ u) {
  __shared__ __align__(16) h16 wl[HPA * 32];   // 12 KB weight k-tile
  int tid  = threadIdx.x;
  int wid  = tid >> 5;
  int lane = tid & 31;
  int blk  = blockIdx.x;
  int gt4  = blk % (GT / 4);
  int rest = blk / (GT / 4);
  int r    = rest % 3;
  int bn   = rest / 3;
  int gt   = gt4 * 4 + wid;
  const h16*   xc   = xh + (size_t)bn * HW * CIN;
  const h16*   wA   = wpa + (size_t)r * HPA * KPA;
  const float* bias = (r == 0) ? b1 : (r == 1) ? b2 : b3;
  v8f acc[12];
#pragma unroll
  for (int i = 0; i < 12; i++) acc[i] = {};
#pragma unroll 1
  for (int ks = 0; ks < 18; ks++) {
    __syncthreads();
    stage_wtile(wA + ks * 32, KPA, HPA, wl, tid);
    __syncthreads();
    v16h bf = loadB_im2col(xc, CIN, ks * 32, gt * 16, r + 1, lane);
#pragma unroll
    for (int mt = 0; mt < 12; mt++) {
      v16h af = loadA_rm(wl, 32, mt * 16, 0, lane);
      acc[mt] = wmma16(af, bf, acc[mt]);
    }
  }
  int col  = gt * 16 + (lane & 15);
  int rowb = 8 * (lane >> 4);
  h16* dst = u + ((size_t)bn * HW + col) * KPA + r * HPA;
#pragma unroll
  for (int mt = 0; mt < 12; mt++) {
    v8h o;
#pragma unroll
    for (int j = 0; j < 8; j++) {
      int m = mt * 16 + rowb + j;
      o[j] = (h16)(acc[mt][j] + bias[m]);
    }
    *(v8h*)(dst + mt * 16 + rowb) = o;
  }
}

// ---------------------------------------------------------------------------
// K2: 1x1 convert1: m1 = relu(Wc1[192x576] @ u + b)   (all pos-major)
// ---------------------------------------------------------------------------
__global__ void __launch_bounds__(128)
k_convert1(const h16* __restrict__ u, const h16* __restrict__ wc1,
           const float* __restrict__ bias, h16* __restrict__ m1) {
  __shared__ __align__(16) h16 wl[HPA * 32];
  int tid  = threadIdx.x;
  int wid  = tid >> 5;
  int lane = tid & 31;
  int gt4  = blockIdx.x % (GT / 4);
  int bn   = blockIdx.x / (GT / 4);
  int gt   = gt4 * 4 + wid;
  const h16* Bsrc = u + (size_t)bn * HW * KPA;
  v8f acc[12];
#pragma unroll
  for (int i = 0; i < 12; i++) acc[i] = {};
#pragma unroll 1
  for (int ks = 0; ks < 18; ks++) {
    __syncthreads();
    stage_wtile(wc1 + ks * 32, KPA, HPA, wl, tid);
    __syncthreads();
    v16h bf = loadB_ctg(Bsrc, KPA, ks * 32, gt * 16, lane);
#pragma unroll
    for (int mt = 0; mt < 12; mt++) {
      v16h af = loadA_rm(wl, 32, mt * 16, 0, lane);
      acc[mt] = wmma16(af, bf, acc[mt]);
    }
  }
  int col  = gt * 16 + (lane & 15);
  int rowb = 8 * (lane >> 4);
  h16* dst = m1 + ((size_t)bn * HW + col) * HPA;
#pragma unroll
  for (int mt = 0; mt < 12; mt++) {
    v8h o;
#pragma unroll
    for (int j = 0; j < 8; j++) {
      int m = mt * 16 + rowb + j;
      o[j] = (h16)fmaxf(acc[mt][j] + bias[m], 0.0f);
    }
    *(v8h*)(dst + mt * 16 + rowb) = o;
  }
}

// ---------------------------------------------------------------------------
// K3: 1x1 convert2 + outer relu -> qkv pos-major [bn][2304][192]
// channels 0..63 = q, 64..127 = k, 128..191 = v
// ---------------------------------------------------------------------------
__global__ void __launch_bounds__(128)
k_convert2(const h16* __restrict__ m1, const h16* __restrict__ wc2,
           const float* __restrict__ bias, h16* __restrict__ qkv) {
  __shared__ __align__(16) h16 wl[HPA * 32];
  int tid  = threadIdx.x;
  int wid  = tid >> 5;
  int lane = tid & 31;
  int gt4  = blockIdx.x % (GT / 4);
  int bn   = blockIdx.x / (GT / 4);
  int gt   = gt4 * 4 + wid;
  const h16* Bsrc = m1 + (size_t)bn * HW * HPA;
  v8f acc[12];
#pragma unroll
  for (int i = 0; i < 12; i++) acc[i] = {};
#pragma unroll 1
  for (int ks = 0; ks < 6; ks++) {
    __syncthreads();
    stage_wtile(wc2 + ks * 32, HPA, HPA, wl, tid);
    __syncthreads();
    v16h bf = loadB_ctg(Bsrc, HPA, ks * 32, gt * 16, lane);
#pragma unroll
    for (int mt = 0; mt < 12; mt++) {
      v16h af = loadA_rm(wl, 32, mt * 16, 0, lane);
      acc[mt] = wmma16(af, bf, acc[mt]);
    }
  }
  int col  = gt * 16 + (lane & 15);
  int rowb = 8 * (lane >> 4);
  h16* dst = qkv + ((size_t)bn * HW + col) * HPA;
#pragma unroll
  for (int mt = 0; mt < 12; mt++) {
    v8h o;
#pragma unroll
    for (int j = 0; j < 8; j++) {
      int m = mt * 16 + rowb + j;
      o[j] = (h16)fmaxf(acc[mt][j] + bias[m], 0.0f);
    }
    *(v8h*)(dst + mt * 16 + rowb) = o;
  }
}

// ---------------------------------------------------------------------------
// K4: kv 3x3 conv 64->64 pad1 + relu -> kvtmp pos-major [bn][t][2304][64]
// ---------------------------------------------------------------------------
__global__ void __launch_bounds__(128)
k_kvconv(const h16* __restrict__ qkv, const h16* __restrict__ wkv,
         const float* __restrict__ bias, h16* __restrict__ kvtmp) {
  __shared__ __align__(16) h16 wl[64 * 32];   // 4 KB weight k-tile
  int tid  = threadIdx.x;
  int wid  = tid >> 5;
  int lane = tid & 31;
  int gt4  = blockIdx.x % (GT / 4);
  int rest = blockIdx.x / (GT / 4);
  int t    = rest % 2;
  int bn   = rest / 2;
  int gt   = gt4 * 4 + wid;
  const h16* xc = qkv + (size_t)bn * HW * HPA + 64 + t * 64;  // channel base in row
  v8f acc[4];
#pragma unroll
  for (int i = 0; i < 4; i++) acc[i] = {};
#pragma unroll 1
  for (int ks = 0; ks < 18; ks++) {
    __syncthreads();
    stage_wtile(wkv + ks * 32, KPA, 64, wl, tid);
    __syncthreads();
    v16h bf = loadB_im2col(xc, HPA, ks * 32, gt * 16, 1, lane);
#pragma unroll
    for (int mt = 0; mt < 4; mt++) {
      v16h af = loadA_rm(wl, 32, mt * 16, 0, lane);
      acc[mt] = wmma16(af, bf, acc[mt]);
    }
  }
  int col  = gt * 16 + (lane & 15);
  int rowb = 8 * (lane >> 4);
  h16* dst = kvtmp + (((size_t)bn * 2 + t) * HW + col) * 64;
#pragma unroll
  for (int mt = 0; mt < 4; mt++) {
    v8h o;
#pragma unroll
    for (int j = 0; j < 8; j++) {
      int m = mt * 16 + rowb + j;
      o[j] = (h16)fmaxf(acc[mt][j] + bias[m], 0.0f);
    }
    *(v8h*)(dst + mt * 16 + rowb) = o;
  }
}

// ---------------------------------------------------------------------------
// K5: 3x3 avg pool -> kbuf token-major [bn][256][64], vbuf ch-major [bn][64][256]
// v additionally scaled by 1/PS^2
// ---------------------------------------------------------------------------
__global__ void k_pool(const h16* __restrict__ kvtmp, h16* __restrict__ kbuf,
                       h16* __restrict__ vbuf) {
  int idx = blockIdx.x * blockDim.x + threadIdx.x;
  if (idx >= NBN * 2 * 64 * KVHW) return;
  int ch    = idx & 63;
  int token = (idx >> 6) & 255;
  int t     = (idx >> 14) & 1;
  int bn    = idx >> 15;
  int py = token >> 4, px = token & 15;
  const h16* src = kvtmp + ((size_t)bn * 2 + t) * HW * 64;
  float s = 0.0f;
#pragma unroll
  for (int dy = 0; dy < 3; dy++)
#pragma unroll
    for (int dx = 0; dx < 3; dx++)
      s += (float)src[((size_t)(py * 3 + dy) * W48 + (px * 3 + dx)) * 64 + ch];
  float val = s * (1.0f / 9.0f);
  if (t) {
    val *= (1.0f / 9.0f);
    vbuf[((size_t)bn * 64 + ch) * KVHW + token] = (h16)val;
  } else {
    kbuf[((size_t)bn * KVHW + token) * 64 + ch] = (h16)val;
  }
}

// ---------------------------------------------------------------------------
// K6: leave-one-out sigmoid attention + residual + relu -> resb pos-major
// One wave per (b, n, gtile); sigmoid score tile relayed through LDS.
// ---------------------------------------------------------------------------
__global__ void __launch_bounds__(32)
k_attn(const h16* __restrict__ qkv, const h16* __restrict__ kbuf,
       const h16* __restrict__ vbuf, const float* __restrict__ scales,
       h16* __restrict__ resb) {
  __shared__ __align__(16) h16 slds[16 * KVHW];   // 8 KB score tile [16 x 256]
  int wave = blockIdx.x;
  int lane = threadIdx.x & 31;
  int gt = wave % GT;
  int t  = wave / GT;
  int n  = t % 6;
  int b  = t / 6;
  if (b >= 2) return;
  int bn = b * 6 + n;

  const h16* qbase = qkv + (size_t)bn * HW * HPA;  // q = channels 0..63
  v16h qa0 = loadA_rm(qbase, HPA, gt * 16, 0, lane);
  v16h qa1 = loadA_rm(qbase, HPA, gt * 16, 32, lane);

  v8f acc[4];
#pragma unroll
  for (int i = 0; i < 4; i++) acc[i] = {};

#pragma unroll 1
  for (int mi = 0; mi < 5; mi++) {
    int m = mi + (mi >= n ? 1 : 0);                // skip self view
    const h16* kb = kbuf + ((size_t)b * 6 + m) * KVHW * 64;
    const h16* vb = vbuf + ((size_t)b * 6 + m) * 64 * KVHW;

    // S = sigmoid(Q K^T * 0.125) -> LDS f16 row-major [16 x 256]
#pragma unroll 1
    for (int nt = 0; nt < 16; nt++) {
      v8f s = {};
      v16h bf0 = loadB_ctg(kb, 64, 0, nt * 16, lane);
      v16h bf1 = loadB_ctg(kb, 64, 32, nt * 16, lane);
      s = wmma16(qa0, bf0, s);
      s = wmma16(qa1, bf1, s);
      int colb = nt * 16 + (lane & 15);
      int rowb = 8 * (lane >> 4);
#pragma unroll
      for (int j = 0; j < 8; j++) {
        float sv = 1.0f / (1.0f + __expf(-0.125f * s[j]));
        slds[(rowb + j) * KVHW + colb] = (h16)sv;
      }
    }
    __builtin_amdgcn_wave_barrier();   // DS in-order per wave; pin the compiler

    // acc += S @ V  (K = 256)
#pragma unroll 1
    for (int ks = 0; ks < 8; ks++) {
      v16h af = loadA_rm(slds, KVHW, 0, ks * 32, lane);
#pragma unroll
      for (int dt = 0; dt < 4; dt++) {
        v16h bf = loadB_ctg(vb, KVHW, ks * 32, dt * 16, lane);
        acc[dt] = wmma16(af, bf, acc[dt]);
      }
    }
    __builtin_amdgcn_wave_barrier();
  }

  // res = relu(attn + scales[n] * xv), pos-major [bn][2304][64]
  float sc = scales[n];
  int coln = lane & 15;
  int rowb = 8 * (lane >> 4);
#pragma unroll
  for (int dt = 0; dt < 4; dt++)
#pragma unroll
    for (int j = 0; j < 8; j++) {
      int d = dt * 16 + coln;
      int g = gt * 16 + rowb + j;
      float xvv = (float)qkv[((size_t)bn * HW + g) * HPA + 128 + d];
      float r = acc[dt][j] + sc * xvv;
      resb[((size_t)bn * HW + g) * 64 + d] = (h16)fmaxf(r, 0.0f);
    }
}

// ---------------------------------------------------------------------------
// K7: fused to_out: relu(W2 @ relu(W1 @ res + b1) + b2) -> fp32 output (NCHW)
// ---------------------------------------------------------------------------
__global__ void __launch_bounds__(128)
k_toout(const h16* __restrict__ resb, const h16* __restrict__ wo1,
        const float* __restrict__ b1, const h16* __restrict__ wo2,
        const float* __restrict__ b2, float* __restrict__ out) {
  __shared__ __align__(16) h16 tl[4][16 * 64];   // [pos][ch] per wave
  int wid  = threadIdx.x >> 5;
  int lane = threadIdx.x & 31;
  int wave = blockIdx.x * 4 + wid;
  int gt = wave % GT;
  int bn = wave / GT;
  if (bn >= NBN) return;
  const h16* Bsrc = resb + (size_t)bn * HW * 64;

  v8f a1[4];
#pragma unroll
  for (int i = 0; i < 4; i++) a1[i] = {};
#pragma unroll
  for (int kk = 0; kk < 2; kk++) {
    v16h bf = loadB_ctg(Bsrc, 64, kk * 32, gt * 16, lane);
    v16h af[4];
#pragma unroll
    for (int mt = 0; mt < 4; mt++)
      af[mt] = loadA_rm(wo1, 64, mt * 16, kk * 32, lane);
#pragma unroll
    for (int mt = 0; mt < 4; mt++)
      a1[mt] = wmma16(af[mt], bf, a1[mt]);
  }
  int coln = lane & 15;
  int rowb = 8 * (lane >> 4);
#pragma unroll
  for (int mt = 0; mt < 4; mt++) {
    v8h o;
#pragma unroll
    for (int j = 0; j < 8; j++) {
      int ch = mt * 16 + rowb + j;
      o[j] = (h16)fmaxf(a1[mt][j] + b1[ch], 0.0f);
    }
    *(v8h*)(&tl[wid][coln * 64 + mt * 16 + rowb]) = o;
  }
  __builtin_amdgcn_wave_barrier();

  v8f a2[4];
#pragma unroll
  for (int i = 0; i < 4; i++) a2[i] = {};
#pragma unroll
  for (int kk = 0; kk < 2; kk++) {
    v16h bf = loadB_ctg(&tl[wid][0], 64, kk * 32, 0, lane);
    v16h af[4];
#pragma unroll
    for (int mt = 0; mt < 4; mt++)
      af[mt] = loadA_rm(wo2, 64, mt * 16, kk * 32, lane);
#pragma unroll
    for (int mt = 0; mt < 4; mt++)
      a2[mt] = wmma16(af[mt], bf, a2[mt]);
  }
  int g = gt * 16 + coln;
#pragma unroll
  for (int mt = 0; mt < 4; mt++)
#pragma unroll
    for (int j = 0; j < 8; j++) {
      int c = mt * 16 + rowb + j;
      out[((size_t)bn * 64 + c) * HW + g] = fmaxf(a2[mt][j] + b2[c], 0.0f);
    }
}

// ---------------------------------------------------------------------------
// host: workspace layout + launches
// ---------------------------------------------------------------------------
extern "C" void kernel_launch(void* const* d_in, const int* in_sizes, int n_in,
                              void* d_out, int out_size, void* d_ws, size_t ws_size,
                              hipStream_t stream) {
  const float* x      = (const float*)d_in[0];
  const float* pa_w1  = (const float*)d_in[1];
  const float* pa_b1  = (const float*)d_in[2];
  const float* pa_w2  = (const float*)d_in[3];
  const float* pa_b2  = (const float*)d_in[4];
  const float* pa_w3  = (const float*)d_in[5];
  const float* pa_b3  = (const float*)d_in[6];
  const float* pa_cw1 = (const float*)d_in[7];
  const float* pa_cb1 = (const float*)d_in[8];
  const float* pa_cw2 = (const float*)d_in[9];
  const float* pa_cb2 = (const float*)d_in[10];
  const float* kv_w   = (const float*)d_in[11];
  const float* kv_b   = (const float*)d_in[12];
  const float* out_w1 = (const float*)d_in[13];
  const float* out_b1 = (const float*)d_in[14];
  const float* out_w2 = (const float*)d_in[15];
  const float* out_b2 = (const float*)d_in[16];
  const float* scales = (const float*)d_in[17];
  float* out = (float*)d_out;

  char* ws = (char*)d_ws;
  size_t off = 0;
  auto bump = [&](size_t elems_h16) {
    h16* p = (h16*)(ws + off);
    off = (off + elems_h16 * sizeof(h16) + 255) & ~(size_t)255;
    return p;
  };

  h16* xh    = bump((size_t)NBN * HW * CIN);        // pos-major
  h16* wpa   = bump((size_t)3 * HPA * KPA);         // tap-major packed
  h16* wc1   = bump((size_t)HPA * KPA);
  h16* wc2   = bump((size_t)HPA * HPA);
  h16* wkv   = bump((size_t)64 * KPA);              // tap-major packed
  h16* wo1   = bump((size_t)64 * 64);
  h16* wo2   = bump((size_t)64 * 64);
  h16* u     = bump((size_t)NBN * HW * KPA);        // pos-major
  h16* m1    = bump((size_t)NBN * HW * HPA);        // pos-major
  h16* qkv   = bump((size_t)NBN * HW * HPA);        // pos-major
  h16* kvtmp = bump((size_t)NBN * 2 * HW * 64);     // pos-major
  h16* kbuf  = bump((size_t)NBN * KVHW * 64);       // token-major
  h16* vbuf  = bump((size_t)NBN * 64 * KVHW);       // channel-major
  h16* resb  = bump((size_t)NBN * HW * 64);         // pos-major
  (void)ws_size; (void)n_in; (void)in_sizes; (void)out_size;

  auto cvt = [&](const float* s, h16* d, int n) {
    k_cvt_f16<<<(n + 255) / 256, 256, 0, stream>>>(s, d, n);
  };

  // staging: transpose x, repack 3x3 weights, convert 1x1 weights
  k_transpose_x<<<(NBN * HW * CIN + 255) / 256, 256, 0, stream>>>(x, xh);
  k_pack_w<<<(HPA * KPA + 255) / 256, 256, 0, stream>>>(pa_w1, wpa + 0 * HPA * KPA, HPA);
  k_pack_w<<<(HPA * KPA + 255) / 256, 256, 0, stream>>>(pa_w2, wpa + 1 * HPA * KPA, HPA);
  k_pack_w<<<(HPA * KPA + 255) / 256, 256, 0, stream>>>(pa_w3, wpa + 2 * HPA * KPA, HPA);
  k_pack_w<<<(64 * KPA + 255) / 256, 256, 0, stream>>>(kv_w, wkv, 64);
  cvt(pa_cw1, wc1, HPA * KPA);
  cvt(pa_cw2, wc2, HPA * HPA);
  cvt(out_w1, wo1, 64 * 64);
  cvt(out_w2, wo2, 64 * 64);

  // dilated convs -> u   (blocks share weight tiles in LDS)
  k_dilconv<<<(NBN * 3 * GT) / 4, 128, 0, stream>>>(xh, wpa, pa_b1, pa_b2, pa_b3, u);
  // 1x1 convert1 -> m1
  k_convert1<<<(NBN * GT) / 4, 128, 0, stream>>>(u, wc1, pa_cb1, m1);
  // 1x1 convert2 -> qkv (relu)
  k_convert2<<<(NBN * GT) / 4, 128, 0, stream>>>(m1, wc2, pa_cb2, qkv);
  // kv conv on xk, xv -> kvtmp
  k_kvconv<<<(NBN * 2 * GT) / 4, 128, 0, stream>>>(qkv, wkv, kv_b, kvtmp);
  // pool -> kbuf/vbuf
  k_pool<<<(NBN * 2 * 64 * KVHW) / 256, 256, 0, stream>>>(kvtmp, kbuf, vbuf);
  // attention + residual + relu -> resb
  k_attn<<<2 * 6 * GT, 32, 0, stream>>>(qkv, kbuf, vbuf, scales, resb);
  // fused to_out -> fp32 output
  k_toout<<<(NBN * GT) / 4, 128, 0, stream>>>(resb, wo1, out_b1, wo2, out_b2, out);
}
